// ScaledDotProductAttention_3513283248597
// MI455X (gfx1250) — matrix operand order repaired and checked by reading, not compiled, not run on hardware
//
#include <hip/hip_runtime.h>

typedef __attribute__((ext_vector_type(16))) _Float16 v16h;
typedef __attribute__((ext_vector_type(8)))  _Float16 v8h;
typedef __attribute__((ext_vector_type(8)))  float    v8f;

#define B_    16
#define S_    4096
#define D_    64
#define WAVES 8        // waves per block
#define QTILE 16       // query rows per wave
#define KTILE 64       // keys per main-loop iteration

union AFrag { v16h v; v8h h[2]; };

__device__ __forceinline__ v8f wmma_f16(v16h a, v16h b, v8f c) {
  // (neg_a, A, neg_b, B, c_mod, C, reuse_a, reuse_b)
  return __builtin_amdgcn_wmma_f32_16x16x32_f16(false, a, false, b, (short)0, c,
                                                false, false);
}

// ---------------- pre-pass: fp32 -> f16 (scale folded into Q) ----------------
__global__ void cvt_qk_kernel(const float* __restrict__ q,
                              const float* __restrict__ k,
                              _Float16* __restrict__ qh,
                              _Float16* __restrict__ kh, int n) {
  int i = blockIdx.x * blockDim.x + threadIdx.x;
  if (i < n) {
    qh[i] = (_Float16)(q[i] * 0.125f);   // 1/sqrt(64)
    kh[i] = (_Float16)(k[i]);
  }
}

// V [B][S][D] fp32 -> Vt [B][D][S] f16 (transposed so PV B-operand is contiguous)
__global__ void cvt_v_kernel(const float* __restrict__ v,
                             _Float16* __restrict__ vt) {
  int idx = blockIdx.x * blockDim.x + threadIdx.x;   // over B*S*D
  int d = idx % D_;
  int s = (idx / D_) % S_;
  int b = idx / (D_ * S_);
  vt[((size_t)(b * D_ + d)) * S_ + s] = (_Float16)v[idx];
}

// ---------------- flash-attention forward (wave32 WMMA) ----------------
__global__ __launch_bounds__(WAVES * 32)
void attn_fwd(const _Float16* __restrict__ Qh, const _Float16* __restrict__ Kh,
              const _Float16* __restrict__ Vt, const int* __restrict__ mask,
              float* __restrict__ Out) {
  __shared__ _Float16 pbuf[WAVES][QTILE * KTILE];   // 16 KB: per-wave P staging

  const int wave = threadIdx.x >> 5;
  const int lane = threadIdx.x & 31;
  const int lrow = lane & 15;      // N (or M) index within 16
  const int g    = lane >> 4;      // half-wave group

  const int tilesPerBatch = S_ / (WAVES * QTILE);        // 32
  const int b      = blockIdx.x / tilesPerBatch;
  const int qc     = blockIdx.x % tilesPerBatch;
  const int q_base = qc * (WAVES * QTILE) + wave * QTILE;

  const v8f vzero = {0.f, 0.f, 0.f, 0.f, 0.f, 0.f, 0.f, 0.f};

  // constant ones B-operand for row-sum-by-matmul
  AFrag ones;
#pragma unroll
  for (int i = 0; i < 2; ++i)
#pragma unroll
    for (int j = 0; j < 8; ++j) ones.h[i][j] = (_Float16)1.0f;

  // ---- Q fragments (A-layout, two K=32 chunks covering D=64) ----
  const _Float16* qrow = Qh + ((size_t)(b * S_ + q_base + lrow)) * D_;
  AFrag qa[2];
#pragma unroll
  for (int c = 0; c < 2; ++c) {
    qa[c].h[0] = *(const v8h*)(qrow + c * 32 + 16 * g);
    qa[c].h[1] = *(const v8h*)(qrow + c * 32 + 16 * g + 8);
  }

  v8f o[4];
  float m[8], l[8];
#pragma unroll
  for (int n = 0; n < 4; ++n) o[n] = vzero;
#pragma unroll
  for (int r = 0; r < 8; ++r) { m[r] = -1e30f; l[r] = 0.f; }

  const _Float16* kbase = Kh + (size_t)b * S_ * D_;
  const _Float16* vbase = Vt + (size_t)b * D_ * S_;
  _Float16* pw = &pbuf[wave][0];

  for (int k0 = 0; k0 < S_; k0 += KTILE) {
    if (k0 + KTILE < S_) {   // prefetch next K/V tiles (global_prefetch_b8)
      __builtin_prefetch(kbase + (size_t)(k0 + KTILE + lane) * D_, 0, 3);
      __builtin_prefetch(kbase + (size_t)(k0 + KTILE + 32 + lane) * D_, 0, 3);
      __builtin_prefetch(vbase + (size_t)lane * S_ + k0 + KTILE, 0, 3);
      __builtin_prefetch(vbase + (size_t)(32 + lane) * S_ + k0 + KTILE, 0, 3);
    }

    // ---- mask bias as C-init: (mask-1)*1e4 -> 0 (keep) / -10000 (masked) ----
    v8f cb[4];
#pragma unroll
    for (int t = 0; t < 4; ++t)
#pragma unroll
      for (int r = 0; r < 8; ++r) {
        int mk = mask[(size_t)(q_base + r + 8 * g) * S_ + k0 + t * 16 + lrow];
        cb[t][r] = (float)(mk - 1) * 10000.0f;
      }

    // ---- scores: two passes of 4 INDEPENDENT WMMAs (no D->C back-to-back) ----
    AFrag kb[4];
    v8f s[4];
#pragma unroll
    for (int t = 0; t < 4; ++t) {   // d-chunk 0: d = [16g, 16g+16)
      const _Float16* krow = kbase + (size_t)(k0 + t * 16 + lrow) * D_;
      kb[t].h[0] = *(const v8h*)(krow + 16 * g);
      kb[t].h[1] = *(const v8h*)(krow + 16 * g + 8);
    }
#pragma unroll
    for (int t = 0; t < 4; ++t) s[t] = wmma_f16(qa[0].v, kb[t].v, cb[t]);
#pragma unroll
    for (int t = 0; t < 4; ++t) {   // d-chunk 1: d = 32 + [16g, 16g+16)
      const _Float16* krow = kbase + (size_t)(k0 + t * 16 + lrow) * D_;
      kb[t].h[0] = *(const v8h*)(krow + 32 + 16 * g);
      kb[t].h[1] = *(const v8h*)(krow + 32 + 16 * g + 8);
    }
#pragma unroll
    for (int t = 0; t < 4; ++t) s[t] = wmma_f16(qa[1].v, kb[t].v, s[t]);

    // ---- online max (shuffle tree), exp, stage P in LDS ----
#pragma unroll
    for (int r = 0; r < 8; ++r) {
      float mx = fmaxf(fmaxf(s[0][r], s[1][r]), fmaxf(s[2][r], s[3][r]));
#pragma unroll
      for (int off = 1; off < 16; off <<= 1)
        mx = fmaxf(mx, __shfl_xor(mx, off, 32));
      float mnew = fmaxf(m[r], mx);
      float corr = __expf(m[r] - mnew);
      m[r] = mnew;
      l[r] *= corr;                       // row-sum added later via WMMA
#pragma unroll
      for (int n = 0; n < 4; ++n) o[n][r] *= corr;
      int prow = (r + 8 * g) * KTILE;
#pragma unroll
      for (int t = 0; t < 4; ++t) {
        float p = __expf(s[t][r] - mnew);
        pw[prow + t * 16 + lrow] = (_Float16)p;
      }
    }

    asm volatile("s_wait_dscnt 0" ::: "memory");

    // ---- reload P in A-layout (two K=32 fragments) ----
    AFrag pa[2];
#pragma unroll
    for (int c = 0; c < 2; ++c) {
      pa[c].h[0] = *(const v8h*)(pw + lrow * KTILE + c * 32 + 16 * g);
      pa[c].h[1] = *(const v8h*)(pw + lrow * KTILE + c * 32 + 16 * g + 8);
    }

    // ---- PV + row-sum: c outer so consecutive WMMAs hit distinct accumulators
    v8f rs = wmma_f16(pa[0].v, ones.v, vzero);   // row-sum part 1
#pragma unroll
    for (int n = 0; n < 4; ++n) {                // 4 independent WMMAs, A=pa[0]
      const _Float16* vrow = vbase + (size_t)(n * 16 + lrow) * S_ + k0 + 16 * g;
      AFrag vb;
      vb.h[0] = *(const v8h*)(vrow);
      vb.h[1] = *(const v8h*)(vrow + 8);
      o[n] = wmma_f16(pa[0].v, vb.v, o[n]);
    }
    rs = wmma_f16(pa[1].v, ones.v, rs);          // row-sum part 2 (5 slots later)
#pragma unroll
    for (int n = 0; n < 4; ++n) {                // 4 independent WMMAs, A=pa[1]
      const _Float16* vrow =
          vbase + (size_t)(n * 16 + lrow) * S_ + k0 + 32 + 16 * g;
      AFrag vb;
      vb.h[0] = *(const v8h*)(vrow);
      vb.h[1] = *(const v8h*)(vrow + 8);
      o[n] = wmma_f16(pa[1].v, vb.v, o[n]);
    }
#pragma unroll
    for (int r = 0; r < 8; ++r) l[r] += rs[r];
  }

  // ---- epilogue: normalize and store fp32 ----
#pragma unroll
  for (int r = 0; r < 8; ++r) {
    float inv = 1.0f / l[r];
    int row = q_base + r + 8 * g;
    float* orow = Out + ((size_t)(b * S_ + row)) * D_;
#pragma unroll
    for (int n = 0; n < 4; ++n)
      orow[n * 16 + lrow] = o[n][r] * inv;
  }
}

// ---------------- host side ----------------
extern "C" void kernel_launch(void* const* d_in, const int* in_sizes, int n_in,
                              void* d_out, int out_size, void* d_ws, size_t ws_size,
                              hipStream_t stream) {
  const float* q    = (const float*)d_in[0];
  const float* k    = (const float*)d_in[1];
  const float* v    = (const float*)d_in[2];
  const int*   mask = (const int*)d_in[3];
  float*       out  = (float*)d_out;

  const size_t nqk = (size_t)B_ * S_ * D_;   // 4,194,304 elements
  _Float16* qh = (_Float16*)d_ws;            // 8 MB
  _Float16* kh = qh + nqk;                   // 8 MB
  _Float16* vt = kh + nqk;                   // 8 MB (transposed V)

  const int threads = 256;
  const int blocks  = (int)(nqk / threads);
  cvt_qk_kernel<<<blocks, threads, 0, stream>>>(q, k, qh, kh, (int)nqk);
  cvt_v_kernel<<<blocks, threads, 0, stream>>>(v, vt);

  const int grid = B_ * (S_ / (WAVES * QTILE));   // 512 blocks, 8 waves each
  attn_fwd<<<grid, WAVES * 32, 0, stream>>>(qh, kh, vt, mask, out);
}
